// Baseline1_89309549953835
// MI455X (gfx1250) — matrix operand order repaired and checked by reading, not compiled, not run on hardware
//
#include <hip/hip_runtime.h>
#include <hip/hip_bf16.h>

// CDNA5 / gfx1250 GRU encoder-decoder forward pass.
// - All matmuls via V_WMMA_F32_16X16X32_F16 (f16 in, f32 accum).
// - Weights pre-packed into WMMA B-fragment layout (ISA 7.12.2), L2-resident.
// - Template KT/NT -> immediate vmem offsets, clauses, load pipelining.
// - 2x2 register blocking (4 accumulators) per wave.
// - Hot recurrent activations staged in LDS (ds_load_b128 A-fragments).

typedef __attribute__((ext_vector_type(16))) _Float16 v16h;
typedef __attribute__((ext_vector_type(8)))  _Float16 v8h;
typedef __attribute__((ext_vector_type(8)))  float    v8f;

static constexpr int NB  = 32;      // batch
static constexpr int NS  = 512;     // seq len
static constexpr int NBS = NB * NS;
static constexpr int HIDN = 1024;
static constexpr long YL_OFF = (long)NB * NS * 100;       // 1638400
static constexpr long YC_OFF = YL_OFF + (long)NB * NS;    // 1654784

// ---------------- WMMA helpers ----------------

__device__ __forceinline__ v8f wmma_(v16h a, v16h b, v8f c) {
  return __builtin_amdgcn_wmma_f32_16x16x32_f16(false, a, false, b, (short)0, c,
                                                false, false);
}

// A 16x32 f16 fragment: lane l row m=l&15, halves = two contiguous 8-half runs
// at +8g and +16+8g (g=l>>4).
__device__ __forceinline__ v16h load_afrag(const _Float16* a, int g) {
  v8h lo = *(const v8h*)(a + 8 * g);
  v8h hi = *(const v8h*)(a + 16 + 8 * g);
  v16h r;
#pragma unroll
  for (int i = 0; i < 8; ++i) { r[i] = lo[i]; r[i + 8] = hi[i]; }
  return r;
}

// Single 16x16 tile: Atile already points at the tile's first row.
template<int NT, int KT>
__device__ __forceinline__ v8f wmma_tile1(const _Float16* Atile, long lda,
                                          const _Float16* Bf, int nt, int lane) {
  const int g = lane >> 4, mr = lane & 15;
  const _Float16* ar = Atile + (long)mr * lda;
  const _Float16* bp = Bf + ((long)nt * 32 + lane) * 16;
  v8f acc = {};
#pragma unroll
  for (int kt = 0; kt < KT; ++kt)
    acc = wmma_(load_afrag(ar + kt * 32, g),
                *(const v16h*)(bp + (long)kt * NT * 512), acc);
  return acc;
}

// 2 (M) x 2 (N) tile block: A covers rows [0,32), B tiles nt0 and nt0+1.
template<int NT, int KT>
__device__ __forceinline__ void wmma_tile2x2(const _Float16* A, long lda,
                                             const _Float16* Bf, int nt0, int lane,
                                             v8f& c00, v8f& c01, v8f& c10, v8f& c11) {
  const int g = lane >> 4, mr = lane & 15;
  const _Float16* a0 = A + (long)mr * lda;
  const _Float16* a1 = A + (long)(16 + mr) * lda;
  const _Float16* bp = Bf + ((long)nt0 * 32 + lane) * 16;
#pragma unroll
  for (int kt = 0; kt < KT; ++kt) {
    v16h av0 = load_afrag(a0 + kt * 32, g);
    v16h av1 = load_afrag(a1 + kt * 32, g);
    v16h bv0 = *(const v16h*)(bp + (long)kt * NT * 512);
    v16h bv1 = *(const v16h*)(bp + (long)kt * NT * 512 + 512);
    c00 = wmma_(av0, bv0, c00);
    c01 = wmma_(av0, bv1, c01);
    c10 = wmma_(av1, bv0, c10);
    c11 = wmma_(av1, bv1, c11);
  }
}

__device__ __forceinline__ void store_f32(v8f acc, float* O, int ldo,
                                          int mt, int nt, int lane) {
  const int col  = nt * 16 + (lane & 15);
  const int row0 = mt * 16 + 8 * (lane >> 4);
#pragma unroll
  for (int r = 0; r < 8; ++r) O[(long)(row0 + r) * ldo + col] = acc[r];
}

__device__ __forceinline__ void store_f16_tanh(v8f acc, _Float16* O, int ldo,
                                               int mt, int nt, int lane,
                                               const float* bias, int Nreal) {
  const int col  = nt * 16 + (lane & 15);
  const float bv = (col < Nreal) ? bias[col] : 0.f;
  const int row0 = mt * 16 + 8 * (lane >> 4);
#pragma unroll
  for (int r = 0; r < 8; ++r)
    O[(long)(row0 + r) * ldo + col] = (_Float16)tanhf(acc[r] + bv);
}

__device__ __forceinline__ float sigmoidf_(float x) { return 1.f / (1.f + __expf(-x)); }

// ---------------- Packing / prep ----------------

// Pack row-major W[Nreal, Kreal] into WMMA B fragments (K pad KT*32, N pad NT*16).
__global__ void pack_bfrag(const float* __restrict__ W, _Float16* __restrict__ dst,
                           int Kreal, int Nreal, int NT, int KT) {
  long i = (long)blockIdx.x * blockDim.x + threadIdx.x;
  long total = (long)KT * NT * 512;
  if (i >= total) return;
  int h    = (int)(i & 15);
  int lane = (int)((i >> 4) & 31);
  long fi  = i >> 9;                       // kt*NT + nt
  int nt = (int)(fi % NT);
  int kt = (int)(fi / NT);
  int k = kt * 32 + (h < 8 ? (lane >> 4) * 8 + h : 16 + (lane >> 4) * 8 + (h - 8));
  int n = nt * 16 + (lane & 15);
  float v = (k < Kreal && n < Nreal) ? W[(long)n * Kreal + k] : 0.f;
  dst[i] = (_Float16)v;
}

// Xcat f16 [BS,128] (coarse[0:20] ++ fine[0:100] ++ pad) and numeric half of
// the joint input J[:,512:1024] = tanh(x_num*num_W + num_b).
__global__ void prep_inputs(const float* __restrict__ xf, const float* __restrict__ xc,
                            const float* __restrict__ numW, const float* __restrict__ numB,
                            _Float16* __restrict__ Xcat, _Float16* __restrict__ J) {
  long i = (long)blockIdx.x * blockDim.x + threadIdx.x;
  if (i >= (long)NBS * 512) return;
  int row = (int)(i >> 9);
  int j   = (int)(i & 511);
  float xn = xf[(long)row * 101 + 100];
  J[(long)row * 1024 + 512 + j] = (_Float16)tanhf(xn * numW[j] + numB[j]);
  if (j < 128) {
    float v = 0.f;
    if (j < 20)       v = xc[(long)row * 21 + j];
    else if (j < 120) v = xf[(long)row * 101 + (j - 20)];
    Xcat[(long)row * 128 + j] = (_Float16)v;
  }
}

// Parallel embedding GEMM: one wave per 2x2 tile block, tanh + f16 store.
template<int NT, int KT>
__global__ void gemm_bias_tanh_f16_k(const _Float16* __restrict__ A, long lda,
                                     const _Float16* __restrict__ Bf,
                                     const float* __restrict__ bias, int Nreal,
                                     _Float16* __restrict__ O, long ldo, int Mtiles) {
  int gw   = (int)(((long)blockIdx.x * blockDim.x + threadIdx.x) >> 5);
  int lane = threadIdx.x & 31;
  int jobs = (Mtiles / 2) * (NT / 2);
  if (gw >= jobs) return;                    // wave-uniform exit
  int mp = gw / (NT / 2), np = gw % (NT / 2);
  v8f c00 = {}, c01 = {}, c10 = {}, c11 = {};
  wmma_tile2x2<NT, KT>(A + (long)mp * 32 * lda, lda, Bf, np * 2, lane,
                       c00, c01, c10, c11);
  store_f16_tanh(c00, O, (int)ldo, mp * 2,     np * 2,     lane, bias, Nreal);
  store_f16_tanh(c01, O, (int)ldo, mp * 2,     np * 2 + 1, lane, bias, Nreal);
  store_f16_tanh(c10, O, (int)ldo, mp * 2 + 1, np * 2,     lane, bias, Nreal);
  store_f16_tanh(c11, O, (int)ldo, mp * 2 + 1, np * 2 + 1, lane, bias, Nreal);
}

// ---------------- Encoder recurrence (1 workgroup, 16 waves, LDS-staged) ----------------

__global__ __launch_bounds__(512, 1)
void encoder_rnn(const _Float16* __restrict__ EMB, const _Float16* __restrict__ PK_IH,
                 const _Float16* __restrict__ PK_HH,
                 const float* __restrict__ b_ih, const float* __restrict__ b_hh,
                 const int* __restrict__ steps,
                 float* __restrict__ H32, float* __restrict__ GI,
                 float* __restrict__ GH, float* __restrict__ SHX) {
  extern __shared__ char smemraw[];
  _Float16* sH = (_Float16*)smemraw;       // [32][1024]
  _Float16* sE = sH + 32 * 1024;           // [32][512]
  const int tid = threadIdx.x, lane = tid & 31, wave = tid >> 5;

  for (int i = tid; i < NB * HIDN; i += 512) { sH[i] = (_Float16)0.f; H32[i] = 0.f; }
  __syncthreads();

  for (int t = 0; t < NS; ++t) {
    // Stage emb_t [32,512] into LDS (v8h copies).
    for (int c = tid; c < NB * 512 / 8; c += 512) {
      int b = c >> 6, k = (c & 63) * 8;
      *(v8h*)(sE + b * 512 + k) = *(const v8h*)(EMB + ((long)b * NS + t) * 512 + k);
    }
    __syncthreads();
    // gh = h @ W_hh^T (96 2x2 jobs) ; gi = emb_t @ W_ih^T (96 2x2 jobs)
    for (int job = wave; job < 192; job += 16) {
      v8f c00 = {}, c01 = {}, c10 = {}, c11 = {};
      if (job < 96) {
        wmma_tile2x2<192, 32>(sH, 1024, PK_HH, job * 2, lane, c00, c01, c10, c11);
        store_f32(c00, GH, 3072, 0, job * 2,     lane);
        store_f32(c01, GH, 3072, 0, job * 2 + 1, lane);
        store_f32(c10, GH, 3072, 1, job * 2,     lane);
        store_f32(c11, GH, 3072, 1, job * 2 + 1, lane);
      } else {
        int j = job - 96;
        wmma_tile2x2<192, 16>(sE, 512, PK_IH, j * 2, lane, c00, c01, c10, c11);
        store_f32(c00, GI, 3072, 0, j * 2,     lane);
        store_f32(c01, GI, 3072, 0, j * 2 + 1, lane);
        store_f32(c10, GI, 3072, 1, j * 2,     lane);
        store_f32(c11, GI, 3072, 1, j * 2 + 1, lane);
      }
    }
    __threadfence(); __syncthreads();
    // GRU gates (torch order r,z,n). H32[i] is thread-private across steps.
    for (int i = tid; i < NB * HIDN; i += 512) {
      int b = i >> 10, j = i & 1023;
      const float* gib = GI + (long)b * 3072;
      const float* ghb = GH + (long)b * 3072;
      float r  = sigmoidf_(gib[j] + b_ih[j] + ghb[j] + b_hh[j]);
      float z  = sigmoidf_(gib[1024 + j] + b_ih[1024 + j] + ghb[1024 + j] + b_hh[1024 + j]);
      float in = gib[2048 + j] + b_ih[2048 + j];
      float hn = ghb[2048 + j] + b_hh[2048 + j];
      float n  = tanhf(in + r * hn);
      float hp = (1.f - z) * n + z * H32[i];
      H32[i] = hp; sH[i] = (_Float16)hp;
      if (steps[b] - 1 == t) SHX[i] = hp;
    }
    __syncthreads();
  }
}

// ---------------- Decoder (1 workgroup, 16 waves, LDS-staged) ----------------

__device__ __forceinline__ void heads_gemm(int wave, int lane, const _Float16* sH,
    const _Float16* PK_CO, const _Float16* PK_FI, const _Float16* PK_LE,
    const float* co_b, const float* fi_b, const float* le_b, float* sLG) {
  for (int job = wave; job < 20; job += 16) {
    v8f acc; int mt, nt, nbase, Nreal; const float* bias;
    if (job < 4) {
      mt = job & 1; nt = job >> 1; nbase = 0; bias = co_b; Nreal = 20;
      acc = wmma_tile1<2, 32>(sH + mt * 16 * 1024, 1024, PK_CO, nt, lane);
    } else if (job < 18) {
      int j = job - 4; mt = j & 1; nt = j >> 1; nbase = 32; bias = fi_b; Nreal = 100;
      acc = wmma_tile1<7, 32>(sH + mt * 16 * 1024, 1024, PK_FI, nt, lane);
    } else {
      mt = job - 18; nt = 0; nbase = 144; bias = le_b; Nreal = 1;
      acc = wmma_tile1<1, 32>(sH + mt * 16 * 1024, 1024, PK_LE, nt, lane);
    }
    int col  = nt * 16 + (lane & 15);
    float bv = (col < Nreal) ? bias[col] : 0.f;
    int row0 = mt * 16 + 8 * (lane >> 4);
#pragma unroll
    for (int r = 0; r < 8; ++r)
      sLG[(long)(row0 + r) * 160 + nbase + col] = acc[r] + bv;
  }
}

__global__ __launch_bounds__(512, 1)
void decoder_rnn(const _Float16* __restrict__ PK_CAT, const _Float16* __restrict__ PK_JOINT,
                 const _Float16* __restrict__ PK_IH, const _Float16* __restrict__ PK_HH,
                 const _Float16* __restrict__ PK_CO, const _Float16* __restrict__ PK_FI,
                 const _Float16* __restrict__ PK_LE,
                 const float* __restrict__ cat_b, const float* __restrict__ num_W,
                 const float* __restrict__ num_b, const float* __restrict__ joint_b,
                 const float* __restrict__ b_ih, const float* __restrict__ b_hh,
                 const float* __restrict__ co_b, const float* __restrict__ fi_b,
                 const float* __restrict__ le_b,
                 const float* __restrict__ xf, const int* __restrict__ steps,
                 const float* __restrict__ SHX,
                 float* __restrict__ H32, float* __restrict__ GI,
                 float* __restrict__ GH, float* __restrict__ out) {
  extern __shared__ char smemraw[];
  _Float16* sH = (_Float16*)smemraw;       // [32][1024] hidden state
  _Float16* sJ = sH + 32 * 1024;           // [32][1024] joint input
  _Float16* sE = sJ + 32 * 1024;           // [32][512]  embedding
  _Float16* sX = sE + 32 * 512;            // [32][128]  decoder input
  float* sLG  = (float*)(smemraw + (32 * 1024 + 32 * 1024 + 32 * 512 + 32 * 128) * 2);
  float* sNum = sLG + 32 * 160;            // [32]
  const int tid = threadIdx.x, lane = tid & 31, wave = tid >> 5;

  // h <- step_hx
  for (int i = tid; i < NB * HIDN; i += 512) { float v = SHX[i]; H32[i] = v; sH[i] = (_Float16)v; }
  __syncthreads();

  // Initial heads -> x_step (no outputs emitted).
  heads_gemm(wave, lane, sH, PK_CO, PK_FI, PK_LE, co_b, fi_b, le_b, sLG);
  __syncthreads();
  if (tid < NB) {
    int b = tid;
    const float* lg = sLG + (long)b * 160;
    int amc = 0; float mv = lg[0];
    for (int j = 1; j < 20; ++j)  if (lg[j] > mv)      { mv = lg[j];      amc = j; }
    int amf = 0; mv = lg[32];
    for (int j = 1; j < 100; ++j) if (lg[32 + j] > mv) { mv = lg[32 + j]; amf = j; }
    float yl = sigmoidf_(lg[144]);
    for (int k = 0; k < 128; ++k) sX[b * 128 + k] = (_Float16)0.f;
    sX[b * 128 + amc]      = (_Float16)1.f;
    sX[b * 128 + 20 + amf] = (_Float16)1.f;
    int idx = steps[b] - 1;
    sNum[b] = xf[((long)b * NS + idx) * 101 + 100] + yl;
  }
  __syncthreads();

  for (int t = 0; t < NS; ++t) {
    // P1: numeric embedding half + cat GEMM (16 jobs) + gh GEMM (96 jobs)
    for (int i = tid; i < NB * 512; i += 512) {
      int b = i >> 9, j = i & 511;
      sJ[b * 1024 + 512 + j] = (_Float16)tanhf(sNum[b] * num_W[j] + num_b[j]);
    }
    for (int job = wave; job < 112; job += 16) {
      v8f c00 = {}, c01 = {}, c10 = {}, c11 = {};
      if (job < 16) {
        wmma_tile2x2<32, 4>(sX, 128, PK_CAT, job * 2, lane, c00, c01, c10, c11);
        store_f16_tanh(c00, sJ, 1024, 0, job * 2,     lane, cat_b, 512);
        store_f16_tanh(c01, sJ, 1024, 0, job * 2 + 1, lane, cat_b, 512);
        store_f16_tanh(c10, sJ, 1024, 1, job * 2,     lane, cat_b, 512);
        store_f16_tanh(c11, sJ, 1024, 1, job * 2 + 1, lane, cat_b, 512);
      } else {
        int j = job - 16;
        wmma_tile2x2<192, 32>(sH, 1024, PK_HH, j * 2, lane, c00, c01, c10, c11);
        store_f32(c00, GH, 3072, 0, j * 2,     lane);
        store_f32(c01, GH, 3072, 0, j * 2 + 1, lane);
        store_f32(c10, GH, 3072, 1, j * 2,     lane);
        store_f32(c11, GH, 3072, 1, j * 2 + 1, lane);
      }
    }
    __threadfence(); __syncthreads();
    // P2: joint GEMM -> emb (LDS)
    for (int job = wave; job < 16; job += 16) {
      v8f c00 = {}, c01 = {}, c10 = {}, c11 = {};
      wmma_tile2x2<32, 32>(sJ, 1024, PK_JOINT, job * 2, lane, c00, c01, c10, c11);
      store_f16_tanh(c00, sE, 512, 0, job * 2,     lane, joint_b, 512);
      store_f16_tanh(c01, sE, 512, 0, job * 2 + 1, lane, joint_b, 512);
      store_f16_tanh(c10, sE, 512, 1, job * 2,     lane, joint_b, 512);
      store_f16_tanh(c11, sE, 512, 1, job * 2 + 1, lane, joint_b, 512);
    }
    __syncthreads();
    // P3: gi GEMM (96 jobs)
    for (int job = wave; job < 96; job += 16) {
      v8f c00 = {}, c01 = {}, c10 = {}, c11 = {};
      wmma_tile2x2<192, 16>(sE, 512, PK_IH, job * 2, lane, c00, c01, c10, c11);
      store_f32(c00, GI, 3072, 0, job * 2,     lane);
      store_f32(c01, GI, 3072, 0, job * 2 + 1, lane);
      store_f32(c10, GI, 3072, 1, job * 2,     lane);
      store_f32(c11, GI, 3072, 1, job * 2 + 1, lane);
    }
    __threadfence(); __syncthreads();
    // P4: GRU gates
    for (int i = tid; i < NB * HIDN; i += 512) {
      int b = i >> 10, j = i & 1023;
      const float* gib = GI + (long)b * 3072;
      const float* ghb = GH + (long)b * 3072;
      float r  = sigmoidf_(gib[j] + b_ih[j] + ghb[j] + b_hh[j]);
      float z  = sigmoidf_(gib[1024 + j] + b_ih[1024 + j] + ghb[1024 + j] + b_hh[1024 + j]);
      float in = gib[2048 + j] + b_ih[2048 + j];
      float hn = ghb[2048 + j] + b_hh[2048 + j];
      float n  = tanhf(in + r * hn);
      float hp = (1.f - z) * n + z * H32[i];
      H32[i] = hp; sH[i] = (_Float16)hp;
    }
    __syncthreads();
    // P5: heads
    heads_gemm(wave, lane, sH, PK_CO, PK_FI, PK_LE, co_b, fi_b, le_b, sLG);
    __syncthreads();
    // P6: log_softmax + outputs + next x
    if (tid < NB) {
      int b = tid;
      const float* lg = sLG + (long)b * 160;
      float m = lg[32];
      for (int j = 1; j < 100; ++j) m = fmaxf(m, lg[32 + j]);
      float s = 0.f;
      for (int j = 0; j < 100; ++j) s += __expf(lg[32 + j] - m);
      float lse = m + __logf(s);
      int amf = 0; float bvf = lg[32];
      for (int j = 0; j < 100; ++j) {
        float v = lg[32 + j];
        out[((long)b * NS + t) * 100 + j] = v - lse;
        if (v > bvf) { bvf = v; amf = j; }
      }
      float mc = lg[0];
      for (int j = 1; j < 20; ++j) mc = fmaxf(mc, lg[j]);
      float sc = 0.f;
      for (int j = 0; j < 20; ++j) sc += __expf(lg[j] - mc);
      float lsec = mc + __logf(sc);
      int amc = 0; float bvc = lg[0];
      for (int j = 0; j < 20; ++j) {
        float v = lg[j];
        out[YC_OFF + ((long)b * NS + t) * 20 + j] = v - lsec;
        if (v > bvc) { bvc = v; amc = j; }
      }
      float yl = sigmoidf_(lg[144]);
      out[YL_OFF + (long)b * NS + t] = yl;
      for (int k = 0; k < 128; ++k) sX[b * 128 + k] = (_Float16)0.f;
      sX[b * 128 + amc]      = (_Float16)1.f;
      sX[b * 128 + 20 + amf] = (_Float16)1.f;
      sNum[b] += yl;
    }
    __syncthreads();
  }
}

// ---------------- Host launcher ----------------

extern "C" void kernel_launch(void* const* d_in, const int* in_sizes, int n_in,
                              void* d_out, int out_size, void* d_ws, size_t ws_size,
                              hipStream_t stream) {
  const float* xf    = (const float*)d_in[0];
  const float* xc    = (const float*)d_in[1];
  const float* cat_W = (const float*)d_in[2];
  const float* cat_b = (const float*)d_in[3];
  const float* num_W = (const float*)d_in[4];
  const float* num_b = (const float*)d_in[5];
  const float* jnt_W = (const float*)d_in[6];
  const float* jnt_b = (const float*)d_in[7];
  const float* W_ih  = (const float*)d_in[8];
  const float* W_hh  = (const float*)d_in[9];
  const float* b_ih  = (const float*)d_in[10];
  const float* b_hh  = (const float*)d_in[11];
  const float* co_W  = (const float*)d_in[12];
  const float* co_b  = (const float*)d_in[13];
  const float* fi_W  = (const float*)d_in[14];
  const float* fi_b  = (const float*)d_in[15];
  const float* le_W  = (const float*)d_in[16];
  const float* le_b  = (const float*)d_in[17];
  const int*   steps = (const int*)d_in[18];
  float* out = (float*)d_out;

  char* ws = (char*)d_ws;
  size_t off = 0;
  auto allocB = [&](size_t bytes) { char* p = ws + off; off += (bytes + 255) & ~(size_t)255; return p; };

  _Float16* PK_CAT   = (_Float16*)allocB((size_t)4  * 32  * 512 * 2);
  _Float16* PK_JOINT = (_Float16*)allocB((size_t)32 * 32  * 512 * 2);
  _Float16* PK_IH    = (_Float16*)allocB((size_t)16 * 192 * 512 * 2);
  _Float16* PK_HH    = (_Float16*)allocB((size_t)32 * 192 * 512 * 2);
  _Float16* PK_CO    = (_Float16*)allocB((size_t)32 * 2   * 512 * 2);
  _Float16* PK_FI    = (_Float16*)allocB((size_t)32 * 7   * 512 * 2);
  _Float16* PK_LE    = (_Float16*)allocB((size_t)32 * 1   * 512 * 2);
  _Float16* XCAT     = (_Float16*)allocB((size_t)NBS * 128 * 2);
  _Float16* JENC     = (_Float16*)allocB((size_t)NBS * 1024 * 2);
  _Float16* EMB      = (_Float16*)allocB((size_t)NBS * 512 * 2);
  float*    H32      = (float*)allocB((size_t)NB * HIDN * 4);
  float*    SHX      = (float*)allocB((size_t)NB * HIDN * 4);
  float*    GI       = (float*)allocB((size_t)NB * 3072 * 4);
  float*    GH       = (float*)allocB((size_t)NB * 3072 * 4);

  auto blocks = [](long n) { return (unsigned)((n + 255) / 256); };

  pack_bfrag<<<blocks((long)4  * 32  * 512), 256, 0, stream>>>(cat_W, PK_CAT, 120, 512, 32, 4);
  pack_bfrag<<<blocks((long)32 * 32  * 512), 256, 0, stream>>>(jnt_W, PK_JOINT, 1024, 512, 32, 32);
  pack_bfrag<<<blocks((long)16 * 192 * 512), 256, 0, stream>>>(W_ih, PK_IH, 512, 3072, 192, 16);
  pack_bfrag<<<blocks((long)32 * 192 * 512), 256, 0, stream>>>(W_hh, PK_HH, 1024, 3072, 192, 32);
  pack_bfrag<<<blocks((long)32 * 2   * 512), 256, 0, stream>>>(co_W, PK_CO, 1024, 20, 2, 32);
  pack_bfrag<<<blocks((long)32 * 7   * 512), 256, 0, stream>>>(fi_W, PK_FI, 1024, 100, 7, 32);
  pack_bfrag<<<blocks((long)32 * 1   * 512), 256, 0, stream>>>(le_W, PK_LE, 1024, 1, 1, 32);

  prep_inputs<<<blocks((long)NBS * 512), 256, 0, stream>>>(xf, xc, num_W, num_b, XCAT, JENC);

  {
    long jobs = (long)(NBS / 16 / 2) * (32 / 2);   // 8192 waves
    gemm_bias_tanh_f16_k<32, 4><<<blocks(jobs * 32), 256, 0, stream>>>(
        XCAT, 128, PK_CAT, cat_b, 512, JENC, 1024, NBS / 16);
    gemm_bias_tanh_f16_k<32, 32><<<blocks(jobs * 32), 256, 0, stream>>>(
        JENC, 1024, PK_JOINT, jnt_b, 512, EMB, 512, NBS / 16);
  }

  const unsigned encLds = (32 * 1024 + 32 * 512) * 2;                       // 96 KB
  const unsigned decLds = (32 * 1024 * 2 + 32 * 512 + 32 * 128) * 2
                          + (32 * 160 + 32) * 4;                            // ~188.1 KB
  encoder_rnn<<<1, 512, encLds, stream>>>(EMB, PK_IH, PK_HH, b_ih, b_hh, steps,
                                          H32, GI, GH, SHX);
  decoder_rnn<<<1, 512, decLds, stream>>>(PK_CAT, PK_JOINT, PK_IH, PK_HH, PK_CO, PK_FI,
                                          PK_LE, cat_b, num_W, num_b, jnt_b, b_ih, b_hh,
                                          co_b, fi_b, le_b, xf, steps, SHX,
                                          H32, GI, GH, out);
  (void)in_sizes; (void)n_in; (void)out_size; (void)ws_size;
}
